// attn_block_70952859730118
// MI455X (gfx1250) — matrix operand order repaired and checked by reading, not compiled, not run on hardware
//
#include <hip/hip_runtime.h>
#include <hip/hip_bf16.h>

typedef _Float16 v8h  __attribute__((ext_vector_type(8)));
typedef _Float16 v16h __attribute__((ext_vector_type(16)));
typedef float    v8f  __attribute__((ext_vector_type(8)));

static __device__ inline v16h cat8(v8h a, v8h b) {
    v16h r;
#pragma unroll
    for (int i = 0; i < 8; ++i) { r[i] = a[i]; r[i + 8] = b[i]; }
    return r;
}

// A-operand fragment (16x32 f16). base points at (row0, k0) of a row-major
// tile with `stride` halves per row. Lane l<16: row l, K {0..7,16..23};
// lane>=16: row l-16, K {8..15,24..31}.
static __device__ inline v16h load_a_frag(const _Float16* base, int stride, int lane) {
    int l = lane & 15, hi = (lane >> 4) & 1;
    const _Float16* p = base + (size_t)l * stride + hi * 8;
    v8h x0 = *(const v8h*)(p);
    v8h x1 = *(const v8h*)(p + 16);
    return cat8(x0, x1);
}

// B-operand fragment (32x16 f16) from an [n][k] tile (stride halves per n-row).
// Lane l<16: col l, K {0..7,16..23}; lane>=16: col l-16, K {8..15,24..31}.
static __device__ inline v16h load_b_frag(const _Float16* base, int stride, int lane) {
    int l = lane & 15, hi = (lane >> 4) & 1;
    const _Float16* p = base + (size_t)l * stride + hi * 8;
    v8h x0 = *(const v8h*)(p);
    v8h x1 = *(const v8h*)(p + 16);
    return cat8(x0, x1);
}

static __device__ inline v8f wmma_f16(v16h a, v16h b, v8f c) {
    return __builtin_amdgcn_wmma_f32_16x16x32_f16(false, a, false, b, (short)0, c,
                                                  false, false);
}

// CDNA5 async DMA: global -> LDS, 16 bytes per lane, tracked by ASYNCcnt.
static __device__ inline void async_load_b128(void* lds_ptr, const void* gptr) {
    unsigned lds_off = (unsigned)(unsigned long long)lds_ptr;
    asm volatile("global_load_async_to_lds_b128 %0, %1, off"
                 :
                 : "v"(lds_off), "v"(gptr)
                 : "memory");
}

static __device__ inline void wait_asynccnt0() {
    asm volatile("s_wait_asynccnt 0" ::: "memory");
}

// ---------------------------------------------------------------------------
// 1) GroupNorm: x[B,256,64,64] -> Ht f16 [B][4096][256]
// ---------------------------------------------------------------------------
__global__ __launch_bounds__(256) void gn_kernel(const float* __restrict__ x,
                                                 const float* __restrict__ sc,
                                                 const float* __restrict__ bs,
                                                 _Float16* __restrict__ Ht) {
    const int g = blockIdx.x;     // 0..31
    const int b = blockIdx.y;     // 0..3
    const int tid = threadIdx.x;
    const size_t xbase = ((size_t)b * 256 + g * 8) * 4096;

    float s = 0.f, ss = 0.f;
    for (int i = tid; i < 8 * 4096; i += 256) {
        float v = x[xbase + i];
        s += v; ss += v * v;
    }
    __shared__ float rs[256], rss[256];
    rs[tid] = s; rss[tid] = ss;
    __syncthreads();
    for (int o = 128; o > 0; o >>= 1) {
        if (tid < o) { rs[tid] += rs[tid + o]; rss[tid] += rss[tid + o]; }
        __syncthreads();
    }
    const float mu  = rs[0] * (1.0f / 32768.0f);
    const float var = rss[0] * (1.0f / 32768.0f) - mu * mu;
    const float rinv = rsqrtf(var + 1e-6f);

    for (int i = tid; i < 8 * 4096; i += 256) {
        int cl = i >> 12, n = i & 4095;
        int c = g * 8 + cl;
        float v = (x[xbase + i] - mu) * rinv * sc[c] + bs[c];
        Ht[((size_t)b * 4096 + n) * 256 + c] = (_Float16)v;
    }
}

// ---------------------------------------------------------------------------
// 2) cast weights fp32 -> f16 (4 x [256][256], kept row-major [c][d])
// ---------------------------------------------------------------------------
__global__ void cast_w_kernel(const float* __restrict__ Wq, const float* __restrict__ Wk,
                              const float* __restrict__ Wv, const float* __restrict__ Wp,
                              _Float16* __restrict__ W16) {
    int i = blockIdx.x * blockDim.x + threadIdx.x;
    if (i < 65536) {
        W16[i]           = (_Float16)Wq[i];
        W16[65536 + i]   = (_Float16)Wk[i];
        W16[131072 + i]  = (_Float16)Wv[i];
        W16[196608 + i]  = (_Float16)Wp[i];
    }
}

// ---------------------------------------------------------------------------
// 3) WMMA GEMM: Out[M,N] = A[M,K] * B[K,N] + bias[col]
//    mode 0: store f16 row-major [row][col]      (Q, K projections)
//    mode 1: store f16 col-major [col][row]      (V projection -> [C][N])
//    mode 2: store f32 col-major + residual      (final proj -> d_out [B,C,H,W])
//    Block tile 128x64, BK=32, 8 waves (4x2), wave tile 32x32.
//    A tile staged via async global->LDS DMA.
// ---------------------------------------------------------------------------
__global__ __launch_bounds__(256) void gemm_kernel(
    const _Float16* __restrict__ A, const _Float16* __restrict__ B,
    const float* __restrict__ bias,
    _Float16* __restrict__ outH, float* __restrict__ outF,
    const float* __restrict__ resid,
    int M, int N, int K, int mode,
    size_t strideA, size_t strideOut, size_t strideRes) {
    __shared__ __align__(16) _Float16 As[128 * 40];
    __shared__ __align__(16) _Float16 Bs[64 * 40];

    const int tid = threadIdx.x, lane = tid & 31, wid = tid >> 5;
    const int wm = wid >> 1, wn = wid & 1;
    const int m0 = blockIdx.y * 128, n0 = blockIdx.x * 64;
    const int bz = blockIdx.z;
    A += (size_t)bz * strideA;

    v8f z = {0, 0, 0, 0, 0, 0, 0, 0};
    v8f acc[4] = {z, z, z, z};
    const int lcol = lane & 15, hi = lane >> 4;

    for (int k0 = 0; k0 < K; k0 += 32) {
        // A tile (128x32): straight copy, async DMA into LDS
#pragma unroll
        for (int it = 0; it < 2; ++it) {
            int c = tid + it * 256;
            int row = c >> 2, kc = (c & 3) * 8;
            async_load_b128(&As[row * 40 + kc],
                            A + (size_t)(m0 + row) * K + k0 + kc);
        }
        // B tile (32x64) transposed into [n][k] via VGPRs
        {
            int kr = tid >> 3, nc = (tid & 7) * 8;
            v8h v = *(const v8h*)(B + (size_t)(k0 + kr) * N + n0 + nc);
#pragma unroll
            for (int i = 0; i < 8; ++i) Bs[(nc + i) * 40 + kr] = v[i];
        }
        if (k0 + 32 < K)
            __builtin_prefetch(A + (size_t)(m0 + (tid >> 1)) * K + k0 + 32, 0, 1);
        wait_asynccnt0();
        __syncthreads();

        v16h af0 = load_a_frag(&As[(wm * 32 + 0) * 40], 40, lane);
        v16h af1 = load_a_frag(&As[(wm * 32 + 16) * 40], 40, lane);
        v16h bf0 = load_b_frag(&Bs[(wn * 32 + 0) * 40], 40, lane);
        v16h bf1 = load_b_frag(&Bs[(wn * 32 + 16) * 40], 40, lane);
        acc[0] = wmma_f16(af0, bf0, acc[0]);
        acc[1] = wmma_f16(af0, bf1, acc[1]);
        acc[2] = wmma_f16(af1, bf0, acc[2]);
        acc[3] = wmma_f16(af1, bf1, acc[3]);
        __syncthreads();
    }

    if (mode == 2) {
        outF  += (size_t)bz * strideOut;
        resid += (size_t)bz * strideRes;
    } else {
        outH += (size_t)bz * strideOut;
    }
#pragma unroll
    for (int mt = 0; mt < 2; ++mt)
#pragma unroll
        for (int nt = 0; nt < 2; ++nt) {
            v8f a = acc[mt * 2 + nt];
#pragma unroll
            for (int r = 0; r < 8; ++r) {
                int row = m0 + wm * 32 + mt * 16 + r + hi * 8;
                int col = n0 + wn * 32 + nt * 16 + lcol;
                float v = a[r] + bias[col];
                if (mode == 0)
                    outH[(size_t)row * N + col] = (_Float16)v;
                else if (mode == 1)
                    outH[(size_t)col * M + row] = (_Float16)v;
                else
                    outF[(size_t)col * M + row] = v + resid[(size_t)col * M + row];
            }
        }
}

// ---------------------------------------------------------------------------
// 4) Flash attention. Qt,Kt: [B][4096][256] f16; Vc: [B][256][4096] f16.
//    Out At: [B][4096][256] f16. Block = 64 query rows, 4 waves x 16 rows.
//    K/V tiles double-buffered via async global->LDS DMA: next tile's DMA
//    overlaps current tile's 64 WMMAs.
// ---------------------------------------------------------------------------
__global__ __launch_bounds__(128) void flash_kernel(const _Float16* __restrict__ Qt,
                                                    const _Float16* __restrict__ Kt,
                                                    const _Float16* __restrict__ Vc,
                                                    _Float16* __restrict__ At) {
    __shared__ __align__(16) _Float16 Ks[2][64 * 264];   // [j][c]
    __shared__ __align__(16) _Float16 Vs[2][256 * 72];   // [d][j]
    __shared__ __align__(16) _Float16 Ps[64 * 72];       // [i][j]

    const int tid = threadIdx.x, lane = tid & 31, wid = tid >> 5;
    const int lcol = lane & 15, hi = lane >> 4;
    const int b = blockIdx.y, q0 = blockIdx.x * 64;
    const size_t bs = (size_t)4096 * 256;
    const _Float16* Qb = Qt + (size_t)b * bs;
    const _Float16* Kb = Kt + (size_t)b * bs;
    const _Float16* Vb = Vc + (size_t)b * bs;

    auto issue_tile = [&](int j0, int buf) {
        for (int c = tid; c < 2048; c += 128) {       // K: 64 rows x 256 halves
            int row = c >> 5, ck = (c & 31) * 8;
            async_load_b128(&Ks[buf][row * 264 + ck],
                            Kb + (size_t)(j0 + row) * 256 + ck);
        }
        for (int c = tid; c < 2048; c += 128) {       // V: 256 rows x 64 halves
            int d = c >> 3, cj = (c & 7) * 8;
            async_load_b128(&Vs[buf][d * 72 + cj],
                            Vb + (size_t)d * 4096 + j0 + cj);
        }
    };

    // Q fragments held in registers: 8 k-steps of 16x32 for this wave's rows.
    issue_tile(0, 0);
    v16h qf[8];
#pragma unroll
    for (int kk = 0; kk < 8; ++kk)
        qf[kk] = load_a_frag(Qb + (size_t)(q0 + wid * 16) * 256 + kk * 32, 256, lane);

    float m[8], l[8];
#pragma unroll
    for (int r = 0; r < 8; ++r) { m[r] = -3.0e38f; l[r] = 0.f; }
    v8f z = {0, 0, 0, 0, 0, 0, 0, 0};
    v8f O[16];
#pragma unroll
    for (int t = 0; t < 16; ++t) O[t] = z;

    const float invs = 0.0625f;  // C^-0.5 = 1/16

    for (int j0 = 0; j0 < 4096; j0 += 64) {
        const int cur = (j0 >> 6) & 1;
        wait_asynccnt0();        // this wave's DMA into buffer `cur` complete
        __syncthreads();         // everyone done with buffer cur^1 and DMA done
        if (j0 + 64 < 4096)
            issue_tile(j0 + 64, cur ^ 1);   // overlap next DMA with compute

        const _Float16* Kc = Ks[cur];
        const _Float16* Vt = Vs[cur];

        // S = Q * K^T for this wave's 16 rows (4 col-tiles of 16 keys)
        v8f S[4] = {z, z, z, z};
#pragma unroll
        for (int kk = 0; kk < 8; ++kk) {
#pragma unroll
            for (int nt = 0; nt < 4; ++nt) {
                v16h bf = load_b_frag(&Kc[(nt * 16) * 264 + kk * 32], 264, lane);
                S[nt] = wmma_f16(qf[kk], bf, S[nt]);
            }
        }

        // online softmax per row
#pragma unroll
        for (int r = 0; r < 8; ++r) {
            float v = -3.0e38f;
#pragma unroll
            for (int nt = 0; nt < 4; ++nt) v = fmaxf(v, S[nt][r]);
#pragma unroll
            for (int off = 1; off < 16; off <<= 1)
                v = fmaxf(v, __shfl_xor(v, off, 32));
            v *= invs;
            float mnew = fmaxf(m[r], v);
            float scale = __expf(m[r] - mnew);
            int prow = wid * 16 + r + hi * 8;
            float rsum = 0.f;
#pragma unroll
            for (int nt = 0; nt < 4; ++nt) {
                float p = __expf(S[nt][r] * invs - mnew);
                rsum += p;
                Ps[prow * 72 + nt * 16 + lcol] = (_Float16)p;
            }
#pragma unroll
            for (int off = 1; off < 16; off <<= 1)
                rsum += __shfl_xor(rsum, off, 32);
            l[r] = l[r] * scale + rsum;
            m[r] = mnew;
#pragma unroll
            for (int t = 0; t < 16; ++t) O[t][r] *= scale;
        }

        // O += P * V   (K-dim = 64 keys, 2 steps; 16 d-tiles)
#pragma unroll
        for (int kk = 0; kk < 2; ++kk) {
            v16h pa = load_a_frag(&Ps[(wid * 16) * 72 + kk * 32], 72, lane);
#pragma unroll
            for (int t = 0; t < 16; ++t) {
                v16h bf = load_b_frag(&Vt[(t * 16) * 72 + kk * 32], 72, lane);
                O[t] = wmma_f16(pa, bf, O[t]);
            }
        }
    }

    _Float16* Ab = At + (size_t)b * bs;
#pragma unroll
    for (int r = 0; r < 8; ++r) {
        float inv = 1.0f / l[r];
        int row = q0 + wid * 16 + r + hi * 8;
#pragma unroll
        for (int t = 0; t < 16; ++t)
            Ab[(size_t)row * 256 + t * 16 + lcol] = (_Float16)(O[t][r] * inv);
    }
}

// ---------------------------------------------------------------------------
extern "C" void kernel_launch(void* const* d_in, const int* in_sizes, int n_in,
                              void* d_out, int out_size, void* d_ws, size_t ws_size,
                              hipStream_t stream) {
    const float* x        = (const float*)d_in[0];
    const float* gn_scale = (const float*)d_in[1];
    const float* gn_bias  = (const float*)d_in[2];
    const float* Wq = (const float*)d_in[3];
    const float* bq = (const float*)d_in[4];
    const float* Wk = (const float*)d_in[5];
    const float* bk = (const float*)d_in[6];
    const float* Wv = (const float*)d_in[7];
    const float* bv = (const float*)d_in[8];
    const float* Wp = (const float*)d_in[9];
    const float* bp = (const float*)d_in[10];
    float* out = (float*)d_out;

    const size_t NB = (size_t)4096 * 256;  // per-batch token x channel block
    _Float16* ws = (_Float16*)d_ws;
    _Float16* Ht  = ws;                 // [B][4096][256]
    _Float16* Qt  = Ht + 4 * NB;        // [B][4096][256]
    _Float16* Kt  = Qt + 4 * NB;        // [B][4096][256]
    _Float16* Vc  = Kt + 4 * NB;        // [B][256][4096]
    _Float16* At  = Vc + 4 * NB;        // [B][4096][256]
    _Float16* W16 = At + 4 * NB;        // 4 x [256][256]
    _Float16* Wq16 = W16;
    _Float16* Wk16 = W16 + 65536;
    _Float16* Wv16 = W16 + 131072;
    _Float16* Wp16 = W16 + 196608;

    // 1) GroupNorm -> Ht
    gn_kernel<<<dim3(32, 4), 256, 0, stream>>>(x, gn_scale, gn_bias, Ht);

    // 2) weight cast
    cast_w_kernel<<<256, 256, 0, stream>>>(Wq, Wk, Wv, Wp, W16);

    // 3) projections: M=4096, N=256, K=256; grid (N/64, M/128, B)
    dim3 ggrid(4, 32, 4);
    gemm_kernel<<<ggrid, 256, 0, stream>>>(Ht, Wq16, bq, Qt, nullptr, nullptr,
                                           4096, 256, 256, 0, NB, NB, 0);
    gemm_kernel<<<ggrid, 256, 0, stream>>>(Ht, Wk16, bk, Kt, nullptr, nullptr,
                                           4096, 256, 256, 0, NB, NB, 0);
    gemm_kernel<<<ggrid, 256, 0, stream>>>(Ht, Wv16, bv, Vc, nullptr, nullptr,
                                           4096, 256, 256, 1, NB, NB, 0);

    // 4) flash attention -> At
    flash_kernel<<<dim3(64, 4), 128, 0, stream>>>(Qt, Kt, Vc, At);

    // 5) output projection + bias + residual -> d_out ([B,C,H,W])
    gemm_kernel<<<ggrid, 256, 0, stream>>>(At, Wp16, bp, nullptr, out, x,
                                           4096, 256, 256, 2, NB, NB, NB);
}
